// FaceGeometryEncoder_14070312862134
// MI455X (gfx1250) — matrix operand order repaired and checked by reading, not compiled, not run on hardware
//
#include <hip/hip_runtime.h>

typedef __attribute__((ext_vector_type(16))) _Float16 v16h;
typedef __attribute__((ext_vector_type(8)))  _Float16 v8h;
typedef __attribute__((ext_vector_type(8)))  float    v8f;
typedef __attribute__((ext_vector_type(4)))  unsigned int u32x4;
typedef __attribute__((ext_vector_type(8)))  int i32x8;
typedef __attribute__((ext_vector_type(4)))  int i32x4;

#define NW      8      // waves per block (wave32)
#define SEG     1024   // faces per segment
#define NCH     (SEG / (NW * 16))   // 8 chunks of 128 faces
#define IN_DIM  7
#define H1D     64
#define H2D     128
#define EMBED   256
#define ATT     64

#if __has_builtin(__builtin_amdgcn_tensor_load_to_lds)
#define HAVE_TDM 1
#else
#define HAVE_TDM 0
#endif

// ---- WMMA fragment helpers (CDNA5 16x16x32 f16 layouts, wave32) ----
// A (16xK, K-chunk of 32): lane m=lane&15, half=lane>>4 holds k = half*16 + i.
// B (Kx16): lane n=lane&15, half=lane>>4 holds k = half*16 + i.
// C/D f32: lane n=lane&15, half=lane>>4, VGPR r -> row m = half*8 + r, col n.

__device__ __forceinline__ v16h wmma_ldA_stage(const _Float16* st, int lane) {
  const int m = lane & 15, half = lane >> 4;
  const v8h* p0 = (const v8h*)&st[m * 32 + half * 16];
  const v8h* p1 = (const v8h*)&st[m * 32 + half * 16 + 8];
  return __builtin_shufflevector(*p0, *p1, 0,1,2,3,4,5,6,7,8,9,10,11,12,13,14,15);
}

// Weights pre-swizzled in LDS: element (block bb, lane, i) at [(bb*32+lane)*16+i]
__device__ __forceinline__ v16h wmma_ldB_sw(const _Float16* W, int bb, int lane) {
  const v8h* p = (const v8h*)&W[(bb * 32 + lane) * 16];
  return __builtin_shufflevector(p[0], p[1], 0,1,2,3,4,5,6,7,8,9,10,11,12,13,14,15);
}

__device__ __forceinline__ v8f wmma_f16(v16h a, v16h b, v8f c) {
  return __builtin_amdgcn_wmma_f32_16x16x32_f16(false, a, false, b, (short)0, c, false, false);
}

__device__ __forceinline__ void preswizzleB(const float* __restrict__ G, _Float16* S,
                                            int K, int N, int tid) {
  const int NT = N >> 4;
  const int total = K * N;
  for (int j = tid; j < total; j += 256) {
    const int i    = j & 15;
    const int lane = (j >> 4) & 31;
    const int bb   = j >> 9;
    const int ks   = bb / NT, t = bb % NT;
    const int k = ks * 32 + (lane >> 4) * 16 + i;
    const int c = t * 16 + (lane & 15);
    S[j] = (_Float16)G[k * N + c];
  }
}

#if HAVE_TDM
// TDM 2D tile load: 128 rows x 7 f32, row stride 7 -> contiguous LDS.
// D# per CDNA5 ISA.
__device__ __forceinline__ void tdm_load_faces(const float* gsrc, unsigned ldsOff) {
  const unsigned long long ga = (unsigned long long)(const void*)gsrc;
  u32x4 g0;
  g0[0] = 1u;                                            // count=1, user mode
  g0[1] = ldsOff;                                        // lds byte address
  g0[2] = (unsigned)ga;                                  // global_addr[31:0]
  g0[3] = (unsigned)((ga >> 32) & 0x01ffffffu) | (2u << 30); // addr[56:32] | type=2
  i32x8 g1;
  g1[0] = (int)(2u << 16);        // workgroup_mask=0, data_size=2 (4 bytes)
  g1[1] = (int)(7u << 16);        // atomic_barrier_addr=0 | tensor_dim0=7 (lo16)
  g1[2] = (int)(128u << 16);      // tensor_dim0 hi=0 | tensor_dim1=128 (lo16)
  g1[3] = (int)(7u << 16);        // tensor_dim1 hi=0 | tile_dim0=7
  g1[4] = 128;                    // tile_dim1=128 | tile_dim2=0
  g1[5] = 7;                      // tensor_dim0_stride[31:0]=7
  g1[6] = 0;                      // stride hi | tensor_dim1_stride lo
  g1[7] = 0;
  i32x4 gz = {0, 0, 0, 0};
#if defined(__clang_major__) && (__clang_major__ >= 23)
  i32x8 gz8 = {0, 0, 0, 0, 0, 0, 0, 0};
  __builtin_amdgcn_tensor_load_to_lds(g0, g1, gz, gz, gz8, 0);
#else
  __builtin_amdgcn_tensor_load_to_lds(g0, g1, gz, gz, 0);
#endif
}
#endif

__global__ __launch_bounds__(256)
void face_geom_encoder_fused(const float* __restrict__ faces,
                             const float* __restrict__ W1, const float* __restrict__ b1,
                             const float* __restrict__ W2, const float* __restrict__ b2,
                             const float* __restrict__ W3, const float* __restrict__ b3,
                             const float* __restrict__ Wa1, const float* __restrict__ ba1,
                             const float* __restrict__ Wa2, const float* __restrict__ ba2,
                             float* __restrict__ out)
{
  __shared__ _Float16 sW1f[32 * H1D];        //  4 KB (K padded 7->32)
  __shared__ _Float16 sW2f[H1D * H2D];       // 16 KB
  __shared__ _Float16 sW3f[H2D * EMBED];     // 64 KB
  __shared__ _Float16 sWa1f[EMBED * ATT];    // 32 KB
  __shared__ float sb1[H1D], sb2[H2D], sb3[EMBED], sba1[ATT], sWa2v[ATT];
  __shared__ _Float16 stage[NW * 16 * 32];   //  8 KB per-wave D->A staging
  __shared__ float ldsFaces[2 * 128 * IN_DIM]; // 7 KB double-buffered face tiles
  __shared__ float ldsAcc[NW * EMBED];       //  8 KB
  __shared__ float ldsM[NW], ldsZ[NW];

  const int tid = threadIdx.x;

  // ---- preload weights into LDS (f16, fragment-swizzled) ----
  for (int j = tid; j < 32 * H1D; j += 256) {
    const int i = j & 15, lane = (j >> 4) & 31, bb = j >> 9;   // bb = t (single K-step)
    const int k = (lane >> 4) * 16 + i;
    const int c = bb * 16 + (lane & 15);
    sW1f[j] = (k < IN_DIM) ? (_Float16)W1[k * H1D + c] : (_Float16)0.f;
  }
  preswizzleB(W2, sW2f, H1D, H2D, tid);
  preswizzleB(W3, sW3f, H2D, EMBED, tid);
  preswizzleB(Wa1, sWa1f, EMBED, ATT, tid);
  if (tid < H1D)  sb1[tid] = b1[tid];
  if (tid < H2D)  sb2[tid] = b2[tid];
  if (tid < EMBED) sb3[tid] = b3[tid];
  if (tid < ATT)  { sba1[tid] = ba1[tid]; sWa2v[tid] = Wa2[tid]; }
  __syncthreads();

  const int wave = tid >> 5;
  const int lane = tid & 31;
  const int n    = lane & 15;
  const int half = lane >> 4;
  _Float16* st = &stage[wave * 512];
  const float ba2v = ba2[0];

  const int seg = blockIdx.x;
  const int segBase = seg * SEG;

  // ---- stage first face tile (async TDM if available) ----
#if HAVE_TDM
  if (wave == 0)
    tdm_load_faces(faces + (size_t)segBase * IN_DIM,
                   (unsigned)(unsigned long long)(const void*)&ldsFaces[0]);
#else
  for (int j = tid; j < 128 * IN_DIM; j += 256)
    ldsFaces[j] = faces[(size_t)segBase * IN_DIM + j];
#endif

  // per-wave online-softmax state
  float mw = -3.0e38f, zw = 0.f;
  float accv[16];
  #pragma unroll
  for (int t = 0; t < 16; ++t) accv[t] = 0.f;

  for (int chunk = 0; chunk < NCH; ++chunk) {
    // ---- wait for this chunk's face tile, then prefetch the next one ----
#if HAVE_TDM
    if (wave == 0) __builtin_amdgcn_s_wait_tensorcnt(0);
#endif
    __syncthreads();   // tile (chunk & 1) ready; all waves done with tile ((chunk+1)&1)

    const float* fbc = &ldsFaces[(chunk & 1) * 128 * IN_DIM];

    // ---- layer-1 A fragment from LDS (K padded 7 -> 32, branch-free) ----
    v16h xa;
    {
      const float* frow = &fbc[(wave * 16 + n) * IN_DIM];
      float f[IN_DIM];
      #pragma unroll
      for (int i = 0; i < IN_DIM; ++i) f[i] = frow[i];
      #pragma unroll
      for (int i = 0; i < 16; ++i) {
        // half==0 lanes own k = 0..15; only k<7 is live data.
        const float v = (i < IN_DIM && half == 0) ? f[i < IN_DIM ? i : 0] : 0.f;
        xa[i] = (_Float16)v;
      }
    }

    if (chunk + 1 < NCH) {
      const float* gsrc = faces + (size_t)(segBase + (chunk + 1) * 128) * IN_DIM;
      float* fbn = &ldsFaces[((chunk + 1) & 1) * 128 * IN_DIM];
#if HAVE_TDM
      if (wave == 0)
        tdm_load_faces(gsrc, (unsigned)(unsigned long long)(const void*)fbn);
#else
      for (int j = tid; j < 128 * IN_DIM; j += 256) fbn[j] = gsrc[j];
#endif
    }

    const v8f zv = {};

    // ---- GEMM1: h1 = X @ W1  (16x64) ----
    v8f h1d[4];
    #pragma unroll
    for (int t = 0; t < 4; ++t)
      h1d[t] = wmma_f16(xa, wmma_ldB_sw(sW1f, t, lane), zv);

    // ---- GEMM2: h2 = relu(h1+b1) @ W2  (16x128) ----
    v8f h2d[8];
    #pragma unroll
    for (int t = 0; t < 8; ++t) h2d[t] = zv;
    #pragma unroll
    for (int ks = 0; ks < 2; ++ks) {
      #pragma unroll
      for (int tt = 0; tt < 2; ++tt) {
        const int ct = ks * 2 + tt;
        const float bc = sb1[ct * 16 + n];
        #pragma unroll
        for (int r = 0; r < 8; ++r)
          st[(half * 8 + r) * 32 + tt * 16 + n] = (_Float16)fmaxf(h1d[ct][r] + bc, 0.f);
      }
      const v16h af = wmma_ldA_stage(st, lane);
      #pragma unroll
      for (int t = 0; t < 8; ++t)
        h2d[t] = wmma_f16(af, wmma_ldB_sw(sW2f, ks * 8 + t, lane), h2d[t]);
    }

    // ---- GEMM3: h3 = relu(h2+b2) @ W3  (16x256, lives in VGPRs) ----
    v8f h3d[16];
    #pragma unroll
    for (int t = 0; t < 16; ++t) h3d[t] = zv;
    #pragma unroll
    for (int ks = 0; ks < 4; ++ks) {
      #pragma unroll
      for (int tt = 0; tt < 2; ++tt) {
        const int ct = ks * 2 + tt;
        const float bc = sb2[ct * 16 + n];
        #pragma unroll
        for (int r = 0; r < 8; ++r)
          st[(half * 8 + r) * 32 + tt * 16 + n] = (_Float16)fmaxf(h2d[ct][r] + bc, 0.f);
      }
      const v16h af = wmma_ldA_stage(st, lane);
      #pragma unroll
      for (int t = 0; t < 16; ++t)
        h3d[t] = wmma_f16(af, wmma_ldB_sw(sW3f, ks * 16 + t, lane), h3d[t]);
    }
    #pragma unroll
    for (int t = 0; t < 16; ++t) {
      const float bc = sb3[t * 16 + n];
      #pragma unroll
      for (int r = 0; r < 8; ++r) h3d[t][r] += bc;
    }

    // ---- GEMM4: a = h3 @ Wa1  (16x64) ----
    v8f ad[4];
    #pragma unroll
    for (int t = 0; t < 4; ++t) ad[t] = zv;
    #pragma unroll
    for (int ks = 0; ks < 8; ++ks) {
      #pragma unroll
      for (int tt = 0; tt < 2; ++tt) {
        const int ct = ks * 2 + tt;
        #pragma unroll
        for (int r = 0; r < 8; ++r)
          st[(half * 8 + r) * 32 + tt * 16 + n] = (_Float16)h3d[ct][r];
      }
      const v16h af = wmma_ldA_stage(st, lane);
      #pragma unroll
      for (int t = 0; t < 4; ++t)
        ad[t] = wmma_f16(af, wmma_ldB_sw(sWa1f, ks * 4 + t, lane), ad[t]);
    }

    // ---- scores: s[m] = tanh(a+ba1) . Wa2 + ba2 ; rows m = half*8 + r ----
    float sc[8];
    #pragma unroll
    for (int r = 0; r < 8; ++r) sc[r] = 0.f;
    #pragma unroll
    for (int t = 0; t < 4; ++t) {
      const float wv = sWa2v[t * 16 + n];
      const float bc = sba1[t * 16 + n];
      #pragma unroll
      for (int r = 0; r < 8; ++r)
        sc[r] += tanhf(ad[t][r] + bc) * wv;
    }
    #pragma unroll
    for (int msk = 1; msk <= 8; msk <<= 1) {
      #pragma unroll
      for (int r = 0; r < 8; ++r) sc[r] += __shfl_xor(sc[r], msk, 32);
    }
    #pragma unroll
    for (int r = 0; r < 8; ++r) sc[r] += ba2v;

    // ---- online softmax update (per-wave) ----
    float cm = sc[0];
    #pragma unroll
    for (int r = 1; r < 8; ++r) cm = fmaxf(cm, sc[r]);
    cm = fmaxf(cm, __shfl_xor(cm, 16, 32));
    const float nm = fmaxf(mw, cm);
    const float scale = __expf(mw - nm);
    float es[8], zc = 0.f;
    #pragma unroll
    for (int r = 0; r < 8; ++r) { es[r] = __expf(sc[r] - nm); zc += es[r]; }
    zc += __shfl_xor(zc, 16, 32);
    zw = zw * scale + zc;
    #pragma unroll
    for (int t = 0; t < 16; ++t) {
      float p = 0.f;
      #pragma unroll
      for (int r = 0; r < 8; ++r) p += es[r] * h3d[t][r];
      accv[t] = accv[t] * scale + p;     // half-partial (this lane's 8 rows)
    }
    mw = nm;
  }

  // ---- merge the 8 waves' online-softmax partials ----
  #pragma unroll
  for (int t = 0; t < 16; ++t) {
    const float v = accv[t] + __shfl_xor(accv[t], 16, 32);
    if (half == 0) ldsAcc[wave * EMBED + t * 16 + n] = v;
  }
  if (lane == 0) { ldsM[wave] = mw; ldsZ[wave] = zw; }
  __syncthreads();

  float gm = ldsM[0];
  #pragma unroll
  for (int w = 1; w < NW; ++w) gm = fmaxf(gm, ldsM[w]);
  float num = 0.f, den = 0.f;
  #pragma unroll
  for (int w = 0; w < NW; ++w) {
    const float s2 = __expf(ldsM[w] - gm);
    num += ldsAcc[w * EMBED + tid] * s2;
    den += ldsZ[w] * s2;
  }
  out[(size_t)seg * EMBED + tid] = num / den;
}

extern "C" void kernel_launch(void* const* d_in, const int* in_sizes, int n_in,
                              void* d_out, int out_size, void* d_ws, size_t ws_size,
                              hipStream_t stream) {
  (void)in_sizes; (void)n_in; (void)out_size; (void)d_ws; (void)ws_size;
  const float* faces = (const float*)d_in[0];
  const float* W1  = (const float*)d_in[1];
  const float* b1  = (const float*)d_in[2];
  const float* W2  = (const float*)d_in[3];
  const float* b2  = (const float*)d_in[4];
  const float* W3  = (const float*)d_in[5];
  const float* b3  = (const float*)d_in[6];
  const float* Wa1 = (const float*)d_in[7];
  const float* ba1 = (const float*)d_in[8];
  const float* Wa2 = (const float*)d_in[9];
  const float* ba2 = (const float*)d_in[10];
  float* out = (float*)d_out;

  face_geom_encoder_fused<<<dim3(1024), dim3(256), 0, stream>>>(
      faces, W1, b1, W2, b2, W3, b3, Wa1, ba1, Wa2, ba2, out);
}
